// BiLSTMCRF_14018773254547
// MI455X (gfx1250) — matrix operand order repaired and checked by reading, not compiled, not run on hardware
//
#include <hip/hip_runtime.h>

#define DEV __device__ __forceinline__

typedef __attribute__((ext_vector_type(16))) __bf16 v16bf;
typedef __attribute__((ext_vector_type(8)))  __bf16 v8bf;
typedef __attribute__((ext_vector_type(8)))  float  v8f;

static constexpr int Bsz = 256;   // batch
static constexpr int Tn  = 200;   // char_num == word_num
static constexpr int HH  = 256;   // hidden per direction
static constexpr int Hf  = 512;   // bi-hidden
static constexpr int Kt  = 12;    // tag size
static constexpr long MROWS = (long)Bsz * Tn;   // 51200

// ---------------------------------------------------------------- WMMA utils
DEV v8f wmma_bf16(v16bf a, v16bf b, v8f c) {
  return __builtin_amdgcn_wmma_f32_16x16x32_bf16(false, a, false, b, (short)0, c,
                                                 false, false);
}

DEV v8f vzero8() {
  v8f z;
#pragma unroll
  for (int i = 0; i < 8; ++i) z[i] = 0.f;
  return z;
}

// A fragment: 16x32 (MxK) tile from row-major [*, lda] bf16.
// lane<16:  row=lane,    v[0..15]=K(k0+0..15)
// lane>=16: row=lane-16, v[0..15]=K(k0+16..31) (contiguous)
DEV v16bf load_a32(const __bf16* base, int lda, int lane, int k0) {
  const int row = lane & 15, half = lane >> 4;
  const __bf16* p = base + (long)row * lda + k0 + half * 16;
  v8bf lo = *(const v8bf*)p;
  v8bf hi = *(const v8bf*)(p + 8);
  return __builtin_shufflevector(lo, hi, 0,1,2,3,4,5,6,7,8,9,10,11,12,13,14,15);
}

// B fragment: 32x16 (KxN) built from row-major weights W[n][k] (i.e. B = W^T).
// lane<16: col=lane, K=k0+0..15; lane>=16: col=lane-16, K=k0+16..31 (contiguous)
DEV v16bf load_b32(const __bf16* Wtile, int ldw, int lane, int k0) {
  const int col = lane & 15, kb = (lane >> 4) * 16;
  const __bf16* p = Wtile + (long)col * ldw + k0 + kb;
  v8bf lo = *(const v8bf*)p;
  v8bf hi = *(const v8bf*)(p + 8);
  return __builtin_shufflevector(lo, hi, 0,1,2,3,4,5,6,7,8,9,10,11,12,13,14,15);
}

// Row-clamped B fragment: branch-free handling of N not multiple of 16.
// Out-of-range columns load row N-1 (in bounds); their products land only in
// output columns that the store phase skips, so no EXEC manipulation needed.
DEV v16bf load_b32_rowclamp(const __bf16* W, int ldw, int lane, int k0,
                            int nbase, int N) {
  int col = nbase + (lane & 15);
  col = (col < N) ? col : (N - 1);
  const int kb = (lane >> 4) * 16;
  const __bf16* p = W + (long)col * ldw + k0 + kb;
  v8bf lo = *(const v8bf*)p;
  v8bf hi = *(const v8bf*)(p + 8);
  return __builtin_shufflevector(lo, hi, 0,1,2,3,4,5,6,7,8,9,10,11,12,13,14,15);
}

DEV float softplusf(float x) { return (x > 15.f) ? x : log1pf(expf(x)); }
DEV float sigmf(float x) { return 1.f / (1.f + expf(-x)); }

// -------------------------------------------------- generic GEMM: out = act(A W^T + b)
// A [M,K] bf16 row-major (M = 51200, grid.x*128), W [N,K(ldw)] bf16 row-major.
__global__ __launch_bounds__(256)
void k_gemm(const __bf16* __restrict__ A, int lda,
            const __bf16* __restrict__ W, int ldw,
            const float* __restrict__ bias,
            void* __restrict__ outp, int ldc,
            int N, int K, int act, int outf32)
{
  const int lane = threadIdx.x & 31;
  const int wave = threadIdx.x >> 5;
  const int m0 = blockIdx.x * 128 + (wave >> 1) * 32;
  const int n0 = blockIdx.y * 64 + (wave & 1) * 32;

  v8f acc00 = vzero8(), acc01 = vzero8(), acc10 = vzero8(), acc11 = vzero8();
  const __bf16* A0 = A + (long)m0 * lda;
  const __bf16* A1 = A0 + (long)16 * lda;

  for (int k = 0; k < K; k += 32) {
    // speculative prefetch of the A stream 256 elements (8 K-steps) ahead (global_prefetch_b8)
    __builtin_prefetch((const void*)(A0 + ((long)(lane & 15)) * lda + k + 256), 0, 1);
    v16bf a0 = load_a32(A0, lda, lane, k);
    v16bf a1 = load_a32(A1, lda, lane, k);
    v16bf b0 = load_b32_rowclamp(W, ldw, lane, k, n0, N);
    v16bf b1 = load_b32_rowclamp(W, ldw, lane, k, n0 + 16, N);
    acc00 = wmma_bf16(a0, b0, acc00);
    acc01 = wmma_bf16(a0, b1, acc01);
    acc10 = wmma_bf16(a1, b0, acc10);
    acc11 = wmma_bf16(a1, b1, acc11);
  }

  const int half = lane >> 4, c0 = lane & 15;
#pragma unroll
  for (int mi = 0; mi < 2; ++mi) {
#pragma unroll
    for (int ni = 0; ni < 2; ++ni) {
      v8f acc = (mi == 0) ? (ni == 0 ? acc00 : acc01) : (ni == 0 ? acc10 : acc11);
      int col = n0 + ni * 16 + c0;
      if (col < N) {
        float bv = bias ? bias[col] : 0.f;
#pragma unroll
        for (int v = 0; v < 8; ++v) {
          long row = m0 + mi * 16 + v + half * 8;
          float x = acc[v] + bv;
          if (act) x = softplusf(x);
          if (outf32) ((float*)outp)[row * ldc + col] = x;
          else        ((__bf16*)outp)[row * ldc + col] = (__bf16)x;
        }
      }
    }
  }
}

// -------------------------------------------------- persistent packed BiLSTM
// 32 blocks: r = blockIdx.x>>3 selects {char-f, char-b, word-f, word-b},
// bbase = 32-row batch chunk. h (bf16) in LDS, c in VGPRs. Each wave owns 32
// cells (n = w*32..w*32+31) and computes ALL FOUR gate tiles for them, so the
// cell update stays in registers. Gates: g = x_t Wih^T + h Whh^T + b.
__global__ __launch_bounds__(256)
void k_lstm(const __bf16* __restrict__ xc, const __bf16* __restrict__ xw,
            const __bf16* __restrict__ WihCf, const __bf16* __restrict__ WihCb,
            const __bf16* __restrict__ WihWf, const __bf16* __restrict__ WihWb,
            const __bf16* __restrict__ WhhCf, const __bf16* __restrict__ WhhCb,
            const __bf16* __restrict__ WhhWf, const __bf16* __restrict__ WhhWb,
            const float* __restrict__ bCf, const float* __restrict__ bCb,
            const float* __restrict__ bWf, const float* __restrict__ bWb,
            const int* __restrict__ lenc, const int* __restrict__ lenw,
            __bf16* __restrict__ outc, __bf16* __restrict__ outw)
{
  const int r = blockIdx.x >> 3;
  const int bbase = (blockIdx.x & 7) * 32;
  const __bf16* xseq = (r < 2) ? xc : xw;
  const int Kin = (r < 2) ? 32 : 128;
  const __bf16* Wih = (r == 0) ? WihCf : (r == 1) ? WihCb : (r == 2) ? WihWf : WihWb;
  const __bf16* Whh = (r == 0) ? WhhCf : (r == 1) ? WhhCb : (r == 2) ? WhhWf : WhhWb;
  const float* bias = (r == 0) ? bCf : (r == 1) ? bCb : (r == 2) ? bWf : bWb;
  const int* len = (r < 2) ? lenc : lenw;
  __bf16* out = (r < 2) ? outc : outw;
  const int coloff = (r & 1) ? HH : 0;
  const bool rev = (r & 1);

  __shared__ __bf16 hsh[32 * 256];
  const int lane = threadIdx.x & 31;
  const int w = threadIdx.x >> 5;
  const int half = lane >> 4, c0 = lane & 15;

  for (int i = threadIdx.x; i < 32 * 256; i += 256) hsh[i] = (__bf16)0.f;

  v8f creg[2][2];
#pragma unroll
  for (int j = 0; j < 2; ++j)
#pragma unroll
    for (int mi = 0; mi < 2; ++mi) creg[j][mi] = vzero8();

  int lenr[2][8];
#pragma unroll
  for (int mi = 0; mi < 2; ++mi)
#pragma unroll
    for (int v = 0; v < 8; ++v)
      lenr[mi][v] = len[bbase + mi * 16 + v + half * 8];

  float biasr[2][4];
#pragma unroll
  for (int j = 0; j < 2; ++j)
#pragma unroll
    for (int g = 0; g < 4; ++g)
      biasr[j][g] = bias[g * 256 + w * 32 + j * 16 + c0];

  __syncthreads();

  for (int s = 0; s < Tn; ++s) {
    const int t = rev ? (Tn - 1 - s) : s;
    v8f acc[2][2][4];
#pragma unroll
    for (int j = 0; j < 2; ++j)
#pragma unroll
      for (int mi = 0; mi < 2; ++mi)
#pragma unroll
        for (int g = 0; g < 4; ++g) acc[j][mi][g] = vzero8();

    // x_t contribution (A rows from global, time-major [T,B,Kin])
    const __bf16* xb = xseq + ((long)t * Bsz + bbase) * Kin;
    for (int k = 0; k < Kin; k += 32) {
      v16bf a0 = load_a32(xb, Kin, lane, k);
      v16bf a1 = load_a32(xb + (long)16 * Kin, Kin, lane, k);
#pragma unroll
      for (int j = 0; j < 2; ++j)
#pragma unroll
        for (int g = 0; g < 4; ++g) {
          v16bf bf = load_b32(Wih + (long)(g * 256 + w * 32 + j * 16) * Kin, Kin, lane, k);
          acc[j][0][g] = wmma_bf16(a0, bf, acc[j][0][g]);
          acc[j][1][g] = wmma_bf16(a1, bf, acc[j][1][g]);
        }
    }
    // h_{t-1} contribution (A rows from LDS)
    for (int k = 0; k < 256; k += 32) {
      v16bf a0 = load_a32(hsh, 256, lane, k);
      v16bf a1 = load_a32(hsh + 16 * 256, 256, lane, k);
#pragma unroll
      for (int j = 0; j < 2; ++j)
#pragma unroll
        for (int g = 0; g < 4; ++g) {
          v16bf bf = load_b32(Whh + (long)(g * 256 + w * 32 + j * 16) * 256, 256, lane, k);
          acc[j][0][g] = wmma_bf16(a0, bf, acc[j][0][g]);
          acc[j][1][g] = wmma_bf16(a1, bf, acc[j][1][g]);
        }
    }
    __syncthreads();   // everyone done reading h_{t-1}

    // cell update (packed-sequence masking) + write h_t
#pragma unroll
    for (int j = 0; j < 2; ++j) {
      const int ncol = w * 32 + j * 16 + c0;
#pragma unroll
      for (int mi = 0; mi < 2; ++mi) {
#pragma unroll
        for (int v = 0; v < 8; ++v) {
          const int brow = mi * 16 + v + half * 8;
          const int bglob = bbase + brow;
          float gi = acc[j][mi][0][v] + biasr[j][0];
          float gf = acc[j][mi][1][v] + biasr[j][1];
          float gg = acc[j][mi][2][v] + biasr[j][2];
          float go = acc[j][mi][3][v] + biasr[j][3];
          float cold = creg[j][mi][v];
          float cn = sigmf(gf) * cold + sigmf(gi) * tanhf(gg);
          float hn = sigmf(go) * tanhf(cn);
          bool valid = t < lenr[mi][v];
          float hold = (float)hsh[brow * 256 + ncol];
          creg[j][mi][v] = valid ? cn : cold;
          hsh[brow * 256 + ncol] = (__bf16)(valid ? hn : hold);
          out[((long)bglob * Tn + t) * Hf + coloff + ncol] = (__bf16)(valid ? hn : 0.f);
        }
      }
    }
    __syncthreads();   // h_t visible before next step's GEMM
  }
}

// ---------------------------------------------------------------- prep kernels
__global__ void k_cvt_pad(const float* __restrict__ in, __bf16* __restrict__ out,
                          int N, int Kin, int Kout)
{
  long i = (long)blockIdx.x * 256 + threadIdx.x;
  if (i >= (long)N * Kout) return;
  int k = (int)(i % Kout);
  long n = i / Kout;
  out[i] = (k < Kin) ? (__bf16)in[n * Kin + k] : (__bf16)0.f;
}

__global__ void k_embed(const int* __restrict__ ch, const float* __restrict__ emb,
                        __bf16* __restrict__ out)  // [T,B,32] time-major, K-padded
{
  long i = (long)blockIdx.x * 256 + threadIdx.x;
  if (i >= (long)Tn * Bsz * 32) return;
  int k = (int)(i & 31);
  long tb = i >> 5;
  int b = (int)(tb % Bsz);
  int t = (int)(tb / Bsz);
  int idx = ch[(long)b * Tn + t];
  out[i] = (k < 30) ? (__bf16)emb[(long)idx * 30 + k] : (__bf16)0.f;
}

__global__ void k_words_tm(const float* __restrict__ wsrc, __bf16* __restrict__ out)
{ // [B,WN,128] f32 -> [WN,B,128] bf16
  long i = (long)blockIdx.x * 256 + threadIdx.x;
  if (i >= (long)Tn * Bsz * 128) return;
  int k = (int)(i & 127);
  long tb = i >> 7;
  int b = (int)(tb % Bsz);
  int t = (int)(tb / Bsz);
  out[i] = (__bf16)wsrc[((long)b * Tn + t) * 128 + k];
}

__global__ void k_zero_xpad(__bf16* __restrict__ x)   // zero cols 712..735 of xcat
{
  long i = (long)blockIdx.x * 256 + threadIdx.x;
  if (i >= MROWS * 24) return;
  long row = i / 24;
  int k = 712 + (int)(i % 24);
  x[row * 736 + k] = (__bf16)0.f;
}

// ---------------------------------------------------------------- CRF NLL
__global__ __launch_bounds__(32)
void k_crf(const float* __restrict__ logits, const int* __restrict__ tags,
           const int* __restrict__ lens, const float* __restrict__ trans,
           float* __restrict__ partial)
{
  const int b = blockIdx.x;
  const int lane = threadIdx.x;
  __shared__ float tr[144];
  __shared__ float al[12];
  for (int i = lane; i < 144; i += 32) tr[i] = trans[i];
  if (lane < 12) al[lane] = 0.f;
  __syncthreads();
  const int len = lens[b];
  const float* lg = logits + (long)b * Tn * Kt;
  const int* tg = tags + (long)b * Tn;

  // real path score
  float part = 0.f;
  for (int t = lane; t < Tn; t += 32) {
    if (t < len) {
      int cur = tg[t];
      int prev = (t == 0) ? 10 : tg[t - 1];           // START = 10
      part += lg[t * Kt + cur] + tr[prev * Kt + cur];
    }
  }
#pragma unroll
  for (int off = 16; off > 0; off >>= 1) part += __shfl_down(part, off, 32);

  // forward algorithm (alpha0 = 0, first step includes transitions)
  for (int t = 0; t < Tn; ++t) {
    float an = 0.f;
    if (lane < 12) {
      float m = -3.0e30f;
#pragma unroll
      for (int i2 = 0; i2 < 12; ++i2) m = fmaxf(m, al[i2] + tr[i2 * Kt + lane]);
      float ssum = 0.f;
#pragma unroll
      for (int i2 = 0; i2 < 12; ++i2) ssum += expf(al[i2] + tr[i2 * Kt + lane] - m);
      an = m + logf(ssum) + lg[t * Kt + lane];
    }
    __syncthreads();
    if (lane < 12 && t < len) al[lane] = an;
    __syncthreads();
  }
  float v = (lane < 12) ? (al[lane] + tr[lane * Kt + 11]) : -3.0e30f;  // STOP = 11
  float m = v;
#pragma unroll
  for (int off = 16; off > 0; off >>= 1) m = fmaxf(m, __shfl_xor(m, off, 32));
  float e = (lane < 12) ? expf(v - m) : 0.f;
#pragma unroll
  for (int off = 16; off > 0; off >>= 1) e += __shfl_xor(e, off, 32);
  float total = m + logf(e);
  if (lane == 0) {
    int last = tg[len - 1];
    partial[b] = total - (part + tr[last * Kt + 11]);
  }
}

__global__ void k_reduce(const float* __restrict__ partial, float* __restrict__ out)
{
  if (threadIdx.x == 0) {     // fixed-order, deterministic
    float s = 0.f;
    for (int i = 0; i < Bsz; ++i) s += partial[i];
    out[0] = s;
  }
}

// ---------------------------------------------------------------- launch
extern "C" void kernel_launch(void* const* d_in, const int* in_sizes, int n_in,
                              void* d_out, int out_size, void* d_ws, size_t ws_size,
                              hipStream_t stream) {
  (void)in_sizes; (void)n_in; (void)out_size; (void)ws_size;
  const int*   characters = (const int*)d_in[0];
  const float* words      = (const float*)d_in[1];
  const int*   tags       = (const int*)d_in[2];
  const int*   len_char   = (const int*)d_in[3];
  const int*   len_word   = (const int*)d_in[4];
  const float* char_emb   = (const float*)d_in[5];
  const float* char_Wih_f = (const float*)d_in[6];
  const float* char_Whh_f = (const float*)d_in[7];
  const float* char_b_f   = (const float*)d_in[8];
  const float* char_Wih_b = (const float*)d_in[9];
  const float* char_Whh_b = (const float*)d_in[10];
  const float* char_b_b   = (const float*)d_in[11];
  const float* char_lin_W = (const float*)d_in[12];
  const float* char_lin_b = (const float*)d_in[13];
  const float* word_Wih_f = (const float*)d_in[14];
  const float* word_Whh_f = (const float*)d_in[15];
  const float* word_b_f   = (const float*)d_in[16];
  const float* word_Wih_b = (const float*)d_in[17];
  const float* word_Whh_b = (const float*)d_in[18];
  const float* word_b_b   = (const float*)d_in[19];
  const float* word_lin_W = (const float*)d_in[20];
  const float* word_lin_b = (const float*)d_in[21];
  const float* lin1_W     = (const float*)d_in[22];
  const float* lin1_b     = (const float*)d_in[23];
  const float* lin2_W     = (const float*)d_in[24];
  const float* lin2_b     = (const float*)d_in[25];
  const float* tag_W      = (const float*)d_in[26];
  const float* tag_b      = (const float*)d_in[27];
  const float* trans      = (const float*)d_in[28];

  char* ws = (char*)d_ws;
  size_t off = 0;
  auto take = [&](size_t bytes) -> void* {
    void* p = ws + off;
    off += (bytes + 255) & ~(size_t)255;
    return p;
  };
  __bf16* cvec   = (__bf16*)take((size_t)Tn * Bsz * 32 * 2);
  __bf16* wtm    = (__bf16*)take((size_t)Tn * Bsz * 128 * 2);
  __bf16* WihCf  = (__bf16*)take(1024 * 32 * 2);
  __bf16* WihCb  = (__bf16*)take(1024 * 32 * 2);
  __bf16* WihWf  = (__bf16*)take(1024 * 128 * 2);
  __bf16* WihWb  = (__bf16*)take(1024 * 128 * 2);
  __bf16* WhhCf  = (__bf16*)take(1024 * 256 * 2);
  __bf16* WhhCb  = (__bf16*)take(1024 * 256 * 2);
  __bf16* WhhWf  = (__bf16*)take(1024 * 256 * 2);
  __bf16* WhhWb  = (__bf16*)take(1024 * 256 * 2);
  __bf16* WcL    = (__bf16*)take(512 * 512 * 2);
  __bf16* WwL    = (__bf16*)take(200 * 512 * 2);
  __bf16* W1p    = (__bf16*)take(512 * 736 * 2);
  __bf16* W2c    = (__bf16*)take(256 * 512 * 2);
  __bf16* Wtc    = (__bf16*)take(12 * 256 * 2);
  __bf16* char_h = (__bf16*)take((size_t)MROWS * 512 * 2);
  __bf16* word_h = (__bf16*)take((size_t)MROWS * 512 * 2);
  __bf16* xcat   = (__bf16*)take((size_t)MROWS * 736 * 2);
  __bf16* y1     = (__bf16*)take((size_t)MROWS * 512 * 2);
  __bf16* y2     = (__bf16*)take((size_t)MROWS * 256 * 2);
  float*  logit  = (float*)take((size_t)MROWS * 12 * 4);
  float*  part   = (float*)take(Bsz * 4);

  auto cvt = [&](const float* src, __bf16* dst, int N, int Kin, int Kout) {
    long tot = (long)N * Kout;
    k_cvt_pad<<<dim3((unsigned)((tot + 255) / 256)), dim3(256), 0, stream>>>(
        src, dst, N, Kin, Kout);
  };
  cvt(char_Wih_f, WihCf, 1024, 30, 32);
  cvt(char_Wih_b, WihCb, 1024, 30, 32);
  cvt(word_Wih_f, WihWf, 1024, 128, 128);
  cvt(word_Wih_b, WihWb, 1024, 128, 128);
  cvt(char_Whh_f, WhhCf, 1024, 256, 256);
  cvt(char_Whh_b, WhhCb, 1024, 256, 256);
  cvt(word_Whh_f, WhhWf, 1024, 256, 256);
  cvt(word_Whh_b, WhhWb, 1024, 256, 256);
  cvt(char_lin_W, WcL, 512, 512, 512);
  cvt(word_lin_W, WwL, 200, 512, 512);
  cvt(lin1_W, W1p, 512, 712, 736);
  cvt(lin2_W, W2c, 256, 512, 512);
  cvt(tag_W, Wtc, 12, 256, 256);

  k_embed<<<dim3(6400), dim3(256), 0, stream>>>(characters, char_emb, cvec);
  k_words_tm<<<dim3(25600), dim3(256), 0, stream>>>(words, wtm);
  k_zero_xpad<<<dim3(4800), dim3(256), 0, stream>>>(xcat);

  // all 4 packed LSTM recurrences, batch-split: 32 blocks / 32 WGPs
  k_lstm<<<dim3(32), dim3(256), 0, stream>>>(
      cvec, wtm, WihCf, WihCb, WihWf, WihWb, WhhCf, WhhCb, WhhWf, WhhWb,
      char_b_f, char_b_b, word_b_f, word_b_b, len_char, len_word, char_h, word_h);

  // char_map -> xcat[:, 0:512]
  k_gemm<<<dim3(400, 8), dim3(256), 0, stream>>>(char_h, 512, WcL, 512, char_lin_b,
                                                 (void*)xcat, 736, 512, 512, 1, 0);
  // word_map -> xcat[:, 512:712]  (reshape [B,WN,T]->[B,T,WN] is identity: WN==T)
  k_gemm<<<dim3(400, 4), dim3(256), 0, stream>>>(word_h, 512, WwL, 512, word_lin_b,
                                                 (void*)(xcat + 512), 736, 200, 512, 1, 0);
  // lin1 (K padded 712->736 with zeros)
  k_gemm<<<dim3(400, 8), dim3(256), 0, stream>>>(xcat, 736, W1p, 736, lin1_b,
                                                 (void*)y1, 512, 512, 736, 1, 0);
  // lin2
  k_gemm<<<dim3(400, 4), dim3(256), 0, stream>>>(y1, 512, W2c, 512, lin2_b,
                                                 (void*)y2, 256, 256, 512, 1, 0);
  // tag head -> f32 logits
  k_gemm<<<dim3(400, 1), dim3(256), 0, stream>>>(y2, 256, Wtc, 256, tag_b,
                                                 (void*)logit, 12, 12, 256, 1, 1);

  k_crf<<<dim3(Bsz), dim3(32), 0, stream>>>(logit, tags, len_char, trans, part);
  k_reduce<<<dim3(1), dim3(32), 0, stream>>>(part, (float*)d_out);
}
